// LinearAttention_Galerkin_and_Fourier_44942537785554
// MI455X (gfx1250) — matrix operand order repaired and checked by reading, not compiled, not run on hardware
//
#include <hip/hip_runtime.h>
#include <hip/hip_bf16.h>
#include <stdint.h>

// ---------------------------------------------------------------------------
// Types for WMMA fragments (gfx1250 wave32)
// ---------------------------------------------------------------------------
typedef __attribute__((ext_vector_type(16))) _Float16 v16h;
typedef __attribute__((ext_vector_type(8)))  float    v8f;
typedef __attribute__((ext_vector_type(4)))  unsigned int v4u;
typedef __attribute__((ext_vector_type(4)))  float    v4f;
typedef __attribute__((ext_vector_type(4)))  _Float16 v4h;
typedef int v4i_vs __attribute__((vector_size(16)));   // matches builtin param type

union FragH { v16h h; v4u u[2]; };

#define ASTRIDE 40   // 40 halves = 80B row stride: 16B aligned, conflict-free b128 reads
#define BSTRIDE 40

// CDNA5 async global->LDS path (ASYNCcnt), guarded so both toolchains compile.
#if defined(__gfx1250__) && __has_builtin(__builtin_amdgcn_global_load_async_to_lds_b128) && __has_builtin(__builtin_amdgcn_s_wait_asynccnt)
#define USE_ASYNC_LDS 1
#else
#define USE_ASYNC_LDS 0
#endif

#define AS1 __attribute__((address_space(1)))
#define AS3 __attribute__((address_space(3)))

__device__ __forceinline__ void copy16_to_lds(const _Float16* g, _Float16* l) {
#if USE_ASYNC_LDS
    __builtin_amdgcn_global_load_async_to_lds_b128((AS1 v4i_vs*)g, (AS3 v4i_vs*)l, 0, 0);
#else
    *(v4u*)l = *(const v4u*)g;
#endif
}

__device__ __forceinline__ void wait_stage_done() {
#if USE_ASYNC_LDS
    __builtin_amdgcn_s_wait_asynccnt(0);
#endif
}

// ---------------------------------------------------------------------------
// Small helper kernels
// ---------------------------------------------------------------------------
__global__ void cvt4_f32_f16_kernel(const float* __restrict__ in,
                                    _Float16* __restrict__ out, int n4) {
    int i = blockIdx.x * 256 + threadIdx.x;
    if (i < n4) {
        v4f f = ((const v4f*)in)[i];
        v4h h = { (_Float16)f.x, (_Float16)f.y, (_Float16)f.z, (_Float16)f.w };
        ((v4h*)out)[i] = h;
    }
}

__global__ void zero_f32_kernel(float* __restrict__ p, int n) {
    int i = blockIdx.x * 256 + threadIdx.x;
    if (i < n) p[i] = 0.0f;
}

// ---------------------------------------------------------------------------
// Tiled WMMA GEMM:  Out[M,N] = A[M,K] * B[N,K]^T  (A,B f16; B row-major [N][K])
//   block tile 128x64, 8 waves (4x2), each wave 32x32 via 2x2 v_wmma tiles.
//   Double-buffered LDS; next tile staged via async global->LDS while the
//   current tile feeds v_wmma_f32_16x16x32_f16. One barrier per K-step.
// ---------------------------------------------------------------------------
template<bool OUT_F16, bool BIAS>
__global__ __launch_bounds__(256)
void gemm_wmma_kernel(const _Float16* __restrict__ Amat,
                      const _Float16* __restrict__ Bmat,
                      void* __restrict__ Outv,
                      const float* __restrict__ bias,
                      int K, int strideA, int strideOut,
                      long zStrideA, long zStrideB, long zStrideOut)
{
    __shared__ __attribute__((aligned(16))) _Float16 As[2][128 * ASTRIDE];
    __shared__ __attribute__((aligned(16))) _Float16 Bs[2][64 * BSTRIDE];

    const int tid  = threadIdx.x;
    const int lane = tid & 31;
    const int wave = tid >> 5;
    const int wm   = wave >> 1;        // 0..3 (M direction)
    const int wn   = wave & 1;         // 0..1 (N direction)
    const int hi16 = lane >> 4;        // 0: lanes 0-15, 1: lanes 16-31
    const int l16  = lane & 15;

    const int m0 = blockIdx.y * 128;
    const int n0 = blockIdx.x * 64;
    const _Float16* Az = Amat + (long)blockIdx.z * zStrideA;
    const _Float16* Bz = Bmat + (long)blockIdx.z * zStrideB;
    const long zOut = (long)blockIdx.z * zStrideOut;

    const int seg = tid & 3;           // 4 x 16B segments per 32-half row
    const int rowq = tid >> 2;         // 0..63

    auto stage = [&](int buf, int kt) {
        // A tile: 128 rows x 32 halves = 512 x 16B units (2 per thread)
        #pragma unroll
        for (int it = 0; it < 2; ++it) {
            int row = rowq + it * 64;
            copy16_to_lds(Az + (long)(m0 + row) * strideA + kt + seg * 8,
                          &As[buf][row * ASTRIDE + seg * 8]);
        }
        // B tile: 64 rows x 32 halves = 256 x 16B units (1 per thread)
        copy16_to_lds(Bz + (long)(n0 + rowq) * K + kt + seg * 8,
                      &Bs[buf][rowq * BSTRIDE + seg * 8]);
    };

    v8f acc[2][2] = {};
    const int T = K >> 5;

    stage(0, 0);
    wait_stage_done();
    __syncthreads();

    for (int s = 0; s < T; ++s) {
        const int cur = s & 1;
        if (s + 1 < T) stage(cur ^ 1, (s + 1) << 5);   // async fill of next buffer

        // ---- load fragments from current buffer (ISA 7.12.2 16-bit layouts)
        FragH a[2], b[2];
        #pragma unroll
        for (int mt = 0; mt < 2; ++mt) {
            const _Float16* p = &As[cur][(wm * 32 + mt * 16 + l16) * ASTRIDE + hi16 * 8];
            a[mt].u[0] = *(const v4u*)p;           // K = kb..kb+7
            a[mt].u[1] = *(const v4u*)(p + 16);    // K = 16+kb..16+kb+7
        }
        #pragma unroll
        for (int nt = 0; nt < 2; ++nt) {
            const _Float16* p = &Bs[cur][(wn * 32 + nt * 16 + l16) * BSTRIDE + hi16 * 8];
            b[nt].u[0] = *(const v4u*)p;           // K = kb..kb+7
            b[nt].u[1] = *(const v4u*)(p + 16);    // K = 16+kb..16+kb+7
        }
        // ---- 4x v_wmma_f32_16x16x32_f16
        #pragma unroll
        for (int mt = 0; mt < 2; ++mt)
            #pragma unroll
            for (int nt = 0; nt < 2; ++nt)
                acc[mt][nt] = __builtin_amdgcn_wmma_f32_16x16x32_f16(
                    false, a[mt].h, false, b[nt].h,
                    (short)0, acc[mt][nt], false, false);

        if (s + 1 < T) {
            wait_stage_done();
            __syncthreads();
        }
    }

    // ---- epilogue ----
    #pragma unroll
    for (int mt = 0; mt < 2; ++mt) {
        #pragma unroll
        for (int nt = 0; nt < 2; ++nt) {
            int col = n0 + wn * 32 + nt * 16 + l16;
            float bv = 0.0f;
            if (BIAS) bv = bias[col];
            #pragma unroll
            for (int r = 0; r < 8; ++r) {
                int row   = m0 + wm * 32 + mt * 16 + hi16 * 8 + r;
                float val = acc[mt][nt][r] + bv;
                if (OUT_F16)
                    ((_Float16*)Outv)[zOut + (long)row * strideOut + col] = (_Float16)val;
                else
                    ((float*)Outv)[zOut + (long)row * strideOut + col] = val;
            }
        }
    }
}

// ---------------------------------------------------------------------------
// Normalize k,v over head dim (d=32) and write transposed f16:
//   knT[b,h,d,n], vnT[b,h,e,n]  (row-contiguous in n for the dots GEMM)
// ---------------------------------------------------------------------------
__global__ __launch_bounds__(256)
void norm_transpose_kernel(const _Float16* __restrict__ qkvh,
                           _Float16* __restrict__ knT,
                           _Float16* __restrict__ vnT)
{
    int n = blockIdx.x * 256 + threadIdx.x;   // token within batch
    int h = blockIdx.y;
    int b = blockIdx.z;
    long rowbase = ((long)(b * 16384 + n)) * 768;
    const _Float16* kp = qkvh + rowbase + 256 + h * 32;
    const _Float16* vp = qkvh + rowbase + 512 + h * 32;

    float kv[32], vv[32];
    float ks = 0.f, ks2 = 0.f, vs = 0.f, vs2 = 0.f;
    #pragma unroll
    for (int d = 0; d < 32; ++d) {
        float t = (float)kp[d]; kv[d] = t; ks += t; ks2 += t * t;
        float u = (float)vp[d]; vv[d] = u; vs += u; vs2 += u * u;
    }
    float kmu = ks * (1.f / 32.f), vmu = vs * (1.f / 32.f);
    float kr = rsqrtf(ks2 * (1.f / 32.f) - kmu * kmu + 1e-5f);
    float vr = rsqrtf(vs2 * (1.f / 32.f) - vmu * vmu + 1e-5f);

    long tbase = ((long)(b * 8 + h) * 32) * 16384 + n;
    #pragma unroll
    for (int d = 0; d < 32; ++d) {
        knT[tbase + (long)d * 16384] = (_Float16)((kv[d] - kmu) * kr);
        vnT[tbase + (long)d * 16384] = (_Float16)((vv[d] - vmu) * vr);
    }
}

// ---------------------------------------------------------------------------
// dots[b,h,32,32] = sum_n knT[b,h,:,n] * vnT[b,h,:,n]^T  (WMMA reduction over n)
// grid: (32 bh-combos, 64 n-slices), 1 wave per block, atomic f32 finish.
// ---------------------------------------------------------------------------
__global__ __launch_bounds__(32)
void dots_kernel(const _Float16* __restrict__ knT,
                 const _Float16* __restrict__ vnT,
                 float* __restrict__ dots)
{
    int bh    = blockIdx.x;            // 0..31
    int slice = blockIdx.y;            // 0..63
    int lane  = threadIdx.x;
    int hi16  = lane >> 4;
    int l16   = lane & 15;

    const _Float16* kbase = knT + (long)bh * 32 * 16384;
    const _Float16* vbase = vnT + (long)bh * 32 * 16384;

    v8f acc[2][2] = {};
    int n0 = slice * 256;
    for (int ks = 0; ks < 256; ks += 32) {
        int nb = n0 + ks;
        FragH a[2], b[2];
        #pragma unroll
        for (int mt = 0; mt < 2; ++mt) {
            const _Float16* p = kbase + (long)(mt * 16 + l16) * 16384 + nb + hi16 * 8;
            a[mt].u[0] = *(const v4u*)p;
            a[mt].u[1] = *(const v4u*)(p + 16);
        }
        #pragma unroll
        for (int nt = 0; nt < 2; ++nt) {
            const _Float16* p = vbase + (long)(nt * 16 + l16) * 16384 + nb + hi16 * 8;
            b[nt].u[0] = *(const v4u*)p;
            b[nt].u[1] = *(const v4u*)(p + 16);
        }
        #pragma unroll
        for (int mt = 0; mt < 2; ++mt)
            #pragma unroll
            for (int nt = 0; nt < 2; ++nt)
                acc[mt][nt] = __builtin_amdgcn_wmma_f32_16x16x32_f16(
                    false, a[mt].h, false, b[nt].h,
                    (short)0, acc[mt][nt], false, false);
    }

    float* dbh = dots + bh * 1024;
    #pragma unroll
    for (int mt = 0; mt < 2; ++mt)
        #pragma unroll
        for (int nt = 0; nt < 2; ++nt) {
            int e = nt * 16 + l16;
            #pragma unroll
            for (int r = 0; r < 8; ++r) {
                int d = mt * 16 + hi16 * 8 + r;
                atomicAdd(&dbh[d * 32 + e], acc[mt][nt][r]);
            }
        }
}

// ---------------------------------------------------------------------------
// Fold dots with w_out:  tempT[b][o][h*32+d] = (1/n) * sum_e dots[b,h,d,e]*w_out[o,h*32+e]
// ---------------------------------------------------------------------------
__global__ __launch_bounds__(256)
void temp_kernel(const float* __restrict__ dots,
                 const float* __restrict__ w_out,
                 _Float16* __restrict__ tempT)
{
    int o  = blockIdx.x;               // 0..255
    int b  = blockIdx.y;               // 0..3
    int hd = threadIdx.x;              // 0..255
    int h  = hd >> 5;
    int d  = hd & 31;
    const float* dp = dots + ((b * 8 + h) * 32 + d) * 32;
    const float* wp = w_out + o * 256 + h * 32;
    float acc = 0.f;
    #pragma unroll
    for (int e = 0; e < 32; ++e) acc += dp[e] * wp[e];
    tempT[((long)(b * 256 + o)) * 256 + hd] = (_Float16)(acc * (1.0f / 16384.0f));
}

// ---------------------------------------------------------------------------
// Host launcher
// ---------------------------------------------------------------------------
extern "C" void kernel_launch(void* const* d_in, const int* in_sizes, int n_in,
                              void* d_out, int out_size, void* d_ws, size_t ws_size,
                              hipStream_t stream) {
    const float* x      = (const float*)d_in[0];   // [4,16384,256]
    const float* w_qkv  = (const float*)d_in[1];   // [768,256]
    const float* w_out  = (const float*)d_in[2];   // [256,256]
    const float* b_out  = (const float*)d_in[3];   // [256]
    float*       out    = (float*)d_out;           // [4,16384,256]

    char* ws = (char*)d_ws;
    _Float16* xh    = (_Float16*)(ws);                  //  33,554,432 B  [65536][256]
    _Float16* wqkvh = (_Float16*)(ws + 33554432);       //     393,216 B  [768][256]
    _Float16* qkvh  = (_Float16*)(ws + 33947648);       // 100,663,296 B  [65536][768]
    _Float16* knT   = (_Float16*)(ws + 134610944);      //  33,554,432 B  [32][32][16384]
    _Float16* vnT   = (_Float16*)(ws + 168165376);      //  33,554,432 B
    float*    dots  = (float*)   (ws + 201719808);      //     131,072 B  [32][32][32]
    _Float16* tempT = (_Float16*)(ws + 201850880);      //     524,288 B  [4][256][256]

    // 0) prep: x,w_qkv -> f16; zero dots accumulator (deterministic each call)
    cvt4_f32_f16_kernel<<<16384, 256, 0, stream>>>(x, xh, 4194304);
    cvt4_f32_f16_kernel<<<192, 256, 0, stream>>>(w_qkv, wqkvh, 49152);
    zero_f32_kernel<<<128, 256, 0, stream>>>(dots, 32768);

    // 1) qkv = x @ w_qkv^T   (M=65536, N=768, K=256), f16 out
    gemm_wmma_kernel<true, false><<<dim3(12, 512, 1), 256, 0, stream>>>(
        xh, wqkvh, (void*)qkvh, nullptr,
        256, /*strideA=*/256, /*strideOut=*/768,
        0L, 0L, 0L);

    // 2) normalize k,v over d=32 and transpose to [b,h,d|e,n] f16
    norm_transpose_kernel<<<dim3(64, 8, 4), 256, 0, stream>>>(qkvh, knT, vnT);

    // 3) dots = k^T v  per (b,h), WMMA reduction over n with atomic finish
    dots_kernel<<<dim3(32, 64, 1), 32, 0, stream>>>(knT, vnT, dots);

    // 4) fold dots with w_out (and 1/n) into tempT
    temp_kernel<<<dim3(256, 4, 1), 256, 0, stream>>>(dots, w_out, tempT);

    // 5) y = q @ tempT[b]^T + b_out   (per b: M=16384, N=256, K=256), f32 out
    gemm_wmma_kernel<false, true><<<dim3(4, 128, 4), 256, 0, stream>>>(
        qkvh, tempT, (void*)out, b_out,
        256, /*strideA=*/768, /*strideOut=*/256,
        /*zA=*/(long)16384 * 768, /*zB=*/(long)256 * 256, /*zOut=*/(long)16384 * 256);
}
